// GATModel_39694087750195
// MI455X (gfx1250) — matrix operand order repaired and checked by reading, not compiled, not run on hardware
//
#include <hip/hip_runtime.h>
#include <cstdint>
#include <cstddef>

#define GAT_N      100000
#define GAT_E      1600000
#define GAT_M      (GAT_E + GAT_N)   // edges + self loops
#define GAT_INCH   128
#define GAT_OUTCH  64                // HEADS * C
#define GAT_HEADS  8
#define GAT_C      8
#define NEG_SLOPE  0.2f

#define ROWS_PER_BLOCK 32            // 2 row tiles of 16; N/32 = 3125 blocks

typedef __attribute__((ext_vector_type(16))) __bf16   v16bf;
typedef __attribute__((ext_vector_type(8)))  __bf16   v8bf;
typedef __attribute__((ext_vector_type(8)))  float    v8f;
typedef __attribute__((ext_vector_type(4)))  uint32_t u32x4;
typedef __attribute__((ext_vector_type(8)))  uint32_t u32x8;

// ---------------------------------------------------------------------------
// helpers
// ---------------------------------------------------------------------------
__device__ __forceinline__ void edge_sd(const long long* __restrict__ ei,
                                        int e, int& s, int& d) {
  if (e < GAT_E) { s = (int)ei[e]; d = (int)ei[GAT_E + e]; }
  else           { s = d = e - GAT_E; }               // self loop
}

// float atomic max via sign-split int atomics (works with -inf init)
__device__ __forceinline__ void atomicMaxF(float* addr, float val) {
  if (val >= 0.0f)
    atomicMax((int*)addr, __float_as_int(val));
  else
    atomicMin((unsigned int*)addr, __float_as_uint(val));
}

// ---------------------------------------------------------------------------
// TDM: contiguous copy of `nelem` 2-byte elements global -> LDS.
// D# built per CDNA5 ISA (2-group form, <=2D).
// Issued once per wave (EXEC ignored by TENSOR ops); caller waits TENSORcnt.
// ---------------------------------------------------------------------------
__device__ __forceinline__ void tdm_load_contig(uint32_t lds_addr,
                                                uint64_t gaddr,
                                                uint32_t nelem) {
  u32x4 g0;
  g0.x = 1u;                                            // count=1, flags 0
  g0.y = lds_addr;                                      // lds_addr [63:32]
  g0.z = (uint32_t)gaddr;                               // global_addr lo
  g0.w = (uint32_t)((gaddr >> 32) & 0x01FFFFFFu)        // global_addr[56:32]
       | (2u << 30);                                    // type = 2 ("image")
  u32x8 g1;
  g1.s0 = 1u << 16;                                     // wg_mask=0, data_size=1 (2B)
  g1.s1 = (nelem & 0xFFFFu) << 16;                      // tensor_dim0 lo16 @ [63:48]
  g1.s2 = (nelem >> 16) | (1u << 16);                   // tensor_dim0 hi16 | tensor_dim1=1
  g1.s3 = (nelem & 0xFFFFu) << 16;                      // tensor_dim1 hi=0 | tile_dim0
  g1.s4 = 1u;                                           // tile_dim1=1, tile_dim2=0
  g1.s5 = nelem;                                        // tensor_dim0_stride lo32
  g1.s6 = (nelem & 0xFFFFu) << 16;                      // stride0 hi=0 | stride1 lo16
  g1.s7 = (nelem >> 16);                                // stride1 hi32
  asm volatile("tensor_load_to_lds %0, %1" :: "s"(g0), "s"(g1) : "memory");
}

// ---------------------------------------------------------------------------
// K0: precision prep (every call, deterministic):
//   xbf[n][k]  = bf16(x[n][k])           (halves GEMM read bandwidth)
//   wbt[n][k]  = bf16(W[k][n])           (W^T so B fragments are contiguous)
// ---------------------------------------------------------------------------
__global__ void gat_prep(const float* __restrict__ x, const float* __restrict__ W,
                         __bf16* __restrict__ xbf, __bf16* __restrict__ wbt) {
  const long long t = (long long)blockIdx.x * blockDim.x + threadIdx.x;
  if (t < (long long)GAT_N * GAT_INCH) xbf[t] = (__bf16)x[t];
  if (t < (long long)GAT_INCH * GAT_OUTCH) {
    const int k = (int)(t >> 6), n = (int)(t & 63);
    wbt[n * GAT_INCH + k] = (__bf16)W[t];
  }
}

// ---------------------------------------------------------------------------
// K1: h = x @ W with bf16 WMMA (f32 accumulate).
//   block = 128 threads (4 waves); block covers 32 rows x 64 cols.
//   Wave w owns column tile n0 = 16w and loops over 2 row tiles -> 8 WMMAs.
//   LDS tiles staged by the Tensor Data Mover (wave 0), s_wait_tensorcnt.
// ---------------------------------------------------------------------------
__global__ __launch_bounds__(128) void gat_proj_wmma(
    const __bf16* __restrict__ xbf, const __bf16* __restrict__ wbt,
    float* __restrict__ h) {
  __shared__ __attribute__((aligned(32))) __bf16 sX[ROWS_PER_BLOCK * GAT_INCH]; // 8KB  [m][k]
  __shared__ __attribute__((aligned(32))) __bf16 sW[GAT_OUTCH * GAT_INCH];      // 16KB [n][k]

  const int tid  = threadIdx.x;
  const int row0 = blockIdx.x * ROWS_PER_BLOCK;

  if (tid < 32) {   // wave 0 issues both DMA copies
    tdm_load_contig((uint32_t)(uintptr_t)&sX[0],
                    (uint64_t)(uintptr_t)(xbf + (size_t)row0 * GAT_INCH),
                    ROWS_PER_BLOCK * GAT_INCH);
    tdm_load_contig((uint32_t)(uintptr_t)&sW[0],
                    (uint64_t)(uintptr_t)wbt,
                    GAT_OUTCH * GAT_INCH);
    __builtin_amdgcn_s_wait_tensorcnt(0);
  }
  __syncthreads();

  const int lane = tid & 31;
  const int wave = tid >> 5;      // column tile 0..3
  const int half = lane >> 4;     // lane group 0/1
  const int l15  = lane & 15;
  const int n0   = wave * 16;

#pragma unroll
  for (int mt = 0; mt < ROWS_PER_BLOCK / 16; ++mt) {
    v8f acc = {};
#pragma unroll
    for (int ks = 0; ks < GAT_INCH / 32; ++ks) {
      const int kb = ks * 32;
      // A fragment: row m = 16*mt + l15; K = kb + 16*half + {0..15}
      const __bf16* ap = &sX[(mt * 16 + l15) * GAT_INCH + kb + 16 * half];
      const v8bf alo = *(const v8bf*)(ap);
      const v8bf ahi = *(const v8bf*)(ap + 8);
      v16bf a;
#pragma unroll
      for (int i = 0; i < 8; ++i) { a[i] = alo[i]; a[i + 8] = ahi[i]; }
      // B fragment: col n = n0 + l15; K = kb + 16*half + {0..15}
      const v16bf b = *(const v16bf*)&sW[(n0 + l15) * GAT_INCH + kb + 16 * half];

      acc = __builtin_amdgcn_wmma_f32_16x16x32_bf16(
          false, a, false, b, (short)0, acc, false, false);
    }
    // D tile: VGPR r holds row m = r + 8*half, col n = n0 + l15
#pragma unroll
    for (int r = 0; r < 8; ++r) {
      const int m = mt * 16 + r + 8 * half;
      h[(size_t)(row0 + m) * GAT_OUTCH + n0 + l15] = acc[r];
    }
  }
}

// ---------------------------------------------------------------------------
// K2: init seg_max=-inf, seg_sum=0, out=bias (re-run every call: deterministic)
// ---------------------------------------------------------------------------
__global__ void gat_init(float* __restrict__ smax, float* __restrict__ ssum,
                         float* __restrict__ out, const float* __restrict__ bias) {
  const long long t = (long long)blockIdx.x * blockDim.x + threadIdx.x;
  if (t < (long long)GAT_N * GAT_HEADS) {
    smax[t] = __int_as_float(0xFF800000u);  // -inf
    ssum[t] = 0.0f;
  }
  if (t < (long long)GAT_N * GAT_OUTCH) out[t] = bias[t & 63];
}

// ---------------------------------------------------------------------------
// K3: a_src[n,h] = sum_c h[n,h,c]*att_src[h,c];  a_dst likewise
// ---------------------------------------------------------------------------
__global__ void gat_coef(const float* __restrict__ h,
                         const float* __restrict__ att_src,
                         const float* __restrict__ att_dst,
                         float* __restrict__ asrc, float* __restrict__ adst) {
  const long long t = (long long)blockIdx.x * blockDim.x + threadIdx.x;  // n*8 + head
  if (t >= (long long)GAT_N * GAT_HEADS) return;
  const int head = (int)(t & 7);
  const float* hp = h + (size_t)t * GAT_C;   // t*8 == n*64 + head*8
  float s = 0.f, d = 0.f;
#pragma unroll
  for (int c = 0; c < GAT_C; ++c) {
    s = fmaf(hp[c], att_src[head * GAT_C + c], s);
    d = fmaf(hp[c], att_dst[head * GAT_C + c], d);
  }
  asrc[t] = s; adst[t] = d;
}

// ---------------------------------------------------------------------------
// K4: per-edge leaky-relu logits; segment max over dst
// ---------------------------------------------------------------------------
__global__ void gat_logits(const long long* __restrict__ ei,
                           const float* __restrict__ asrc,
                           const float* __restrict__ adst,
                           float* __restrict__ alpha, float* __restrict__ smax) {
  const long long t = (long long)blockIdx.x * blockDim.x + threadIdx.x;
  if (t >= (long long)GAT_M * GAT_HEADS) return;
  const int e = (int)(t >> 3), hd = (int)(t & 7);
  int s, d; edge_sd(ei, e, s, d);
  float lg = asrc[(size_t)s * 8 + hd] + adst[(size_t)d * 8 + hd];
  lg = (lg >= 0.f) ? lg : NEG_SLOPE * lg;
  alpha[t] = lg;
  atomicMaxF(&smax[(size_t)d * 8 + hd], lg);
}

// ---------------------------------------------------------------------------
// K5: ealpha = exp(logit - seg_max[dst]); segment sum over dst
// ---------------------------------------------------------------------------
__global__ void gat_expsum(const long long* __restrict__ ei,
                           const float* __restrict__ smax,
                           float* __restrict__ alpha, float* __restrict__ ssum) {
  const long long t = (long long)blockIdx.x * blockDim.x + threadIdx.x;
  if (t >= (long long)GAT_M * GAT_HEADS) return;
  const int e = (int)(t >> 3), hd = (int)(t & 7);
  int s, d; edge_sd(ei, e, s, d); (void)s;
  const float ea = __expf(alpha[t] - smax[(size_t)d * 8 + hd]);
  alpha[t] = ea;
  atomicAdd(&ssum[(size_t)d * 8 + hd], ea);
}

// ---------------------------------------------------------------------------
// K6: alpha /= (seg_sum[dst] + eps)   (final alpha output, in place in d_out)
// ---------------------------------------------------------------------------
__global__ void gat_norm(const long long* __restrict__ ei,
                         const float* __restrict__ ssum,
                         float* __restrict__ alpha) {
  const long long t = (long long)blockIdx.x * blockDim.x + threadIdx.x;
  if (t >= (long long)GAT_M * GAT_HEADS) return;
  const int e = (int)(t >> 3), hd = (int)(t & 7);
  int s, d; edge_sd(ei, e, s, d); (void)s;
  alpha[t] = alpha[t] / (ssum[(size_t)d * 8 + hd] + 1e-16f);
}

// ---------------------------------------------------------------------------
// K7: out[dst, ch] += alpha[e, ch/8] * h[src, ch]   (64 threads per edge)
// ---------------------------------------------------------------------------
__global__ void gat_scatter(const long long* __restrict__ ei,
                            const float* __restrict__ alpha,
                            const float* __restrict__ h,
                            float* __restrict__ out) {
  const long long t = (long long)blockIdx.x * blockDim.x + threadIdx.x;
  if (t >= (long long)GAT_M * GAT_OUTCH) return;
  const int e  = (int)(t >> 6);
  const int ch = (int)(t & 63);
  const int hd = ch >> 3;
  int s, d; edge_sd(ei, e, s, d);
  const float a = alpha[(size_t)e * 8 + hd];
  atomicAdd(&out[(size_t)d * GAT_OUTCH + ch], a * h[(size_t)s * GAT_OUTCH + ch]);
}

// ---------------------------------------------------------------------------
// launch
// ---------------------------------------------------------------------------
extern "C" void kernel_launch(void* const* d_in, const int* in_sizes, int n_in,
                              void* d_out, int out_size, void* d_ws, size_t ws_size,
                              hipStream_t stream) {
  const float*     x       = (const float*)d_in[0];
  const long long* ei      = (const long long*)d_in[1];   // int64 [2, E]
  const float*     W       = (const float*)d_in[2];
  const float*     att_src = (const float*)d_in[3];
  const float*     att_dst = (const float*)d_in[4];
  const float*     bias    = (const float*)d_in[5];

  float* out   = (float*)d_out;                           // [N, 64]
  float* alpha = out + (size_t)GAT_N * GAT_OUTCH;         // [E+N, 8]

  // workspace layout: h | a_src | a_dst | seg_max | seg_sum | xbf | wbt
  float* h    = (float*)d_ws;
  float* asrc = h    + (size_t)GAT_N * GAT_OUTCH;
  float* adst = asrc + (size_t)GAT_N * GAT_HEADS;
  float* smax = adst + (size_t)GAT_N * GAT_HEADS;
  float* ssum = smax + (size_t)GAT_N * GAT_HEADS;
  __bf16* xbf = (__bf16*)(ssum + (size_t)GAT_N * GAT_HEADS);
  __bf16* wbt = xbf + (size_t)GAT_N * GAT_INCH;

  // 0) bf16 prep: xbf = bf16(x), wbt = bf16(W^T)
  {
    const long long n = (long long)GAT_N * GAT_INCH;
    gat_prep<<<(unsigned)((n + 255) / 256), 256, 0, stream>>>(x, W, xbf, wbt);
  }

  // 1) projection GEMM (WMMA bf16, TDM-staged LDS tiles)
  gat_proj_wmma<<<GAT_N / ROWS_PER_BLOCK, 128, 0, stream>>>(xbf, wbt, h);

  // 2) init accumulators / out=bias
  {
    const long long n = (long long)GAT_N * GAT_OUTCH;
    gat_init<<<(unsigned)((n + 255) / 256), 256, 0, stream>>>(smax, ssum, out, bias);
  }

  // 3) per-node attention coefficients
  {
    const long long n = (long long)GAT_N * GAT_HEADS;
    gat_coef<<<(unsigned)((n + 255) / 256), 256, 0, stream>>>(h, att_src, att_dst, asrc, adst);
  }

  // 4-6) segment softmax over destination nodes
  {
    const long long n = (long long)GAT_M * GAT_HEADS;
    const unsigned g = (unsigned)((n + 255) / 256);
    gat_logits<<<g, 256, 0, stream>>>(ei, asrc, adst, alpha, smax);
    gat_expsum<<<g, 256, 0, stream>>>(ei, smax, alpha, ssum);
    gat_norm  <<<g, 256, 0, stream>>>(ei, ssum, alpha);
  }

  // 7) weighted scatter-aggregate
  {
    const long long n = (long long)GAT_M * GAT_OUTCH;
    gat_scatter<<<(unsigned)((n + 255) / 256), 256, 0, stream>>>(ei, alpha, h, out);
  }
}
